// Mamba4Rec_60911226192343
// MI455X (gfx1250) — matrix operand order repaired and checked by reading, not compiled, not run on hardware
//
// Mamba4Rec forward for MI455X (gfx1250), wave32 + WMMA f16->f32.
// GEMMs: v_wmma_f32_16x16x32_f16, wave tile 32x64 (8 WMMA per K-step),
// branch-free inner loop (out-of-range weight columns are pointer-clamped;
// their garbage accumulator columns are masked at store). Fused epilogues:
// silu->f16, exact-gelu+bias->f16, bias+f32. Selective scan is a
// register-resident recurrence: one lane per (b,d) channel, h[16] in VGPRs.
// Workspace requirement ~520 MB (buffers aliased across pipeline stages).

#include <hip/hip_runtime.h>
#include <hip/hip_bf16.h>
#include <math.h>

typedef __attribute__((ext_vector_type(16))) _Float16 v16h;
typedef __attribute__((ext_vector_type(8)))  _Float16 v8h;
typedef __attribute__((ext_vector_type(8)))  float    v8f;

#define B_   512
#define L_   200
#define D_   128
#define DI_  256
#define DS_  16
#define DTR_ 8
#define FF_  512
#define NL_  2
#define M_   (B_ * L_)   // 102400 rows

// ---------------------------------------------------------------- utilities
__device__ __forceinline__ float wave_sum32(float v) {
#pragma unroll
  for (int off = 16; off > 0; off >>= 1) v += __shfl_xor(v, off, 32);
  return v;
}

__global__ void cvt_f16_kernel(const float* __restrict__ s,
                               _Float16* __restrict__ d, int n) {
  int i = blockIdx.x * 256 + threadIdx.x;
  if (i < n) d[i] = (_Float16)s[i];
}

// ----------------------------------------------------- embedding + LayerNorm
__global__ __launch_bounds__(256)
void embed_ln0_kernel(const int* __restrict__ ids, const float* __restrict__ emb,
                      const float* __restrict__ pos, const float* __restrict__ w,
                      const float* __restrict__ b, float* __restrict__ X,
                      _Float16* __restrict__ XH) {
  int row  = blockIdx.x * 8 + (threadIdx.x >> 5);   // one wave per row
  int lane = threadIdx.x & 31;
  int l    = row % L_;
  int tok  = ids[row];
  float v[4];
  float s = 0.f;
#pragma unroll
  for (int j = 0; j < 4; j++) {
    int c = lane * 4 + j;
    v[j] = emb[(size_t)tok * D_ + c] + pos[l * D_ + c];
    s += v[j];
  }
  float mean = wave_sum32(s) * (1.f / D_);
  float var = 0.f;
#pragma unroll
  for (int j = 0; j < 4; j++) { float dd = v[j] - mean; var += dd * dd; }
  var = wave_sum32(var) * (1.f / D_);
  float r = rsqrtf(var + 1e-12f);
#pragma unroll
  for (int j = 0; j < 4; j++) {
    int c = lane * 4 + j;
    float o = (v[j] - mean) * r * w[c] + b[c];
    X[(size_t)row * D_ + c]  = o;
    XH[(size_t)row * D_ + c] = (_Float16)o;
  }
}

// ------------------------------------------------- residual add + LayerNorm
__global__ __launch_bounds__(256)
void add_ln_kernel(const float* __restrict__ h, const float* __restrict__ xin,
                   const float* __restrict__ w, const float* __restrict__ b,
                   float* __restrict__ Xout, _Float16* __restrict__ XH) {
  int row  = blockIdx.x * 8 + (threadIdx.x >> 5);
  int lane = threadIdx.x & 31;
  float v[4];
  float s = 0.f;
#pragma unroll
  for (int j = 0; j < 4; j++) {
    int c = lane * 4 + j;
    v[j] = h[(size_t)row * D_ + c] + xin[(size_t)row * D_ + c];
    s += v[j];
  }
  float mean = wave_sum32(s) * (1.f / D_);
  float var = 0.f;
#pragma unroll
  for (int j = 0; j < 4; j++) { float dd = v[j] - mean; var += dd * dd; }
  var = wave_sum32(var) * (1.f / D_);
  float r = rsqrtf(var + 1e-12f);
#pragma unroll
  for (int j = 0; j < 4; j++) {
    int c = lane * 4 + j;
    float o = (v[j] - mean) * r * w[c] + b[c];
    Xout[(size_t)row * D_ + c] = o;
    XH[(size_t)row * D_ + c]   = (_Float16)o;
  }
}

// ----------------------------------------------------------- WMMA GEMM core
// C[M,N] = A[M,K] * W[N,K]^T ; A,W f16 row-major, f32 accumulate.
// Wave tile 32(M)x64(N): 2x4 accumulators, 8 WMMA per K-step of 32.
// Block = 8 waves arranged 4(M)x2(N) -> 128x128 block tile.
enum { EPI_F32 = 0, EPI_SILU_F16 = 1, EPI_GELU_BIAS_F16 = 2, EPI_BIAS_F32 = 3 };

__device__ __forceinline__ v16h load_a_frag(const _Float16* p) {
  // per ISA 16-bit A layout: lane<16 -> K 0..7 (VGPR0-3) + K 16..23 (VGPR4-7)
  v8h lo = *(const v8h*)p;
  v8h hi = *(const v8h*)(p + 16);
  return __builtin_shufflevector(lo, hi, 0, 1, 2, 3, 4, 5, 6, 7,
                                 8, 9, 10, 11, 12, 13, 14, 15);
}

template <int EPI>
__global__ __launch_bounds__(256)
void gemm_wmma_kernel(const _Float16* __restrict__ A,
                      const _Float16* __restrict__ W,
                      const float* __restrict__ bias, void* __restrict__ Cv,
                      int Mr, int N, int K) {
  const int lane = threadIdx.x & 31;
  const int wave = threadIdx.x >> 5;
  const int lr   = lane & 15;
  const int hi   = lane >> 4;
  const int m0   = blockIdx.y * 128 + (wave & 3) * 32;
  const int n0   = blockIdx.x * 128 + (wave >> 2) * 64;

  v8f acc[2][4] = {};

  const _Float16* a0p = A + (size_t)(m0 + lr) * K + hi * 8;
  const _Float16* a1p = a0p + (size_t)16 * K;
  // Column-clamped weight pointers: out-of-range columns read row N-1
  // (valid address); their garbage lands in acc columns masked at store.
  const _Float16* bp[4];
#pragma unroll
  for (int t = 0; t < 4; t++) {
    int nb = n0 + t * 16 + lr;
    int nc = nb < N ? nb : (N - 1);
    bp[t] = W + (size_t)nc * K + hi * 8;
  }

  for (int k = 0; k < K; k += 32) {
    v16h a0 = load_a_frag(a0p + k);
    v16h a1 = load_a_frag(a1p + k);
#pragma unroll
    for (int t = 0; t < 4; t++) {
      v16h b = load_a_frag(bp[t] + k);   // B[k][n] = W[n][k]
      acc[0][t] = __builtin_amdgcn_wmma_f32_16x16x32_f16(
          false, a0, false, b, (short)0, acc[0][t], false, false);
      acc[1][t] = __builtin_amdgcn_wmma_f32_16x16x32_f16(
          false, a1, false, b, (short)0, acc[1][t], false, false);
    }
  }

#pragma unroll
  for (int tm = 0; tm < 2; tm++) {
#pragma unroll
    for (int t = 0; t < 4; t++) {
      int col = n0 + t * 16 + lr;
      if (col >= N) continue;
      float bv = 0.f;
      if (EPI == EPI_GELU_BIAS_F16 || EPI == EPI_BIAS_F32) bv = bias[col];
#pragma unroll
      for (int j = 0; j < 8; j++) {
        int row = m0 + tm * 16 + hi * 8 + j;  // C layout: VGPR j -> M = j+8*hi
        size_t idx = (size_t)row * N + col;
        float v = acc[tm][t][j];
        if (EPI == EPI_F32) {
          ((float*)Cv)[idx] = v;
        } else if (EPI == EPI_SILU_F16) {
          float sv = v / (1.f + expf(-v));
          ((_Float16*)Cv)[idx] = (_Float16)sv;
        } else if (EPI == EPI_GELU_BIAS_F16) {
          float tt = v + bv;
          float g = 0.5f * tt * (1.f + erff(tt * 0.70710678118654752f));
          ((_Float16*)Cv)[idx] = (_Float16)g;
        } else {
          ((float*)Cv)[idx] = v + bv;
        }
      }
    }
  }
}

static void launch_gemm(int epi, const _Float16* A, const _Float16* W,
                        const float* bias, void* C, int Mr, int N, int K,
                        hipStream_t s) {
  dim3 g((N + 127) / 128, (Mr + 127) / 128);
  dim3 b(256);
  switch (epi) {
    case EPI_F32:           gemm_wmma_kernel<EPI_F32><<<g, b, 0, s>>>(A, W, bias, C, Mr, N, K); break;
    case EPI_SILU_F16:      gemm_wmma_kernel<EPI_SILU_F16><<<g, b, 0, s>>>(A, W, bias, C, Mr, N, K); break;
    case EPI_GELU_BIAS_F16: gemm_wmma_kernel<EPI_GELU_BIAS_F16><<<g, b, 0, s>>>(A, W, bias, C, Mr, N, K); break;
    default:                gemm_wmma_kernel<EPI_BIAS_F32><<<g, b, 0, s>>>(A, W, bias, C, Mr, N, K); break;
  }
}

// ------------------------------------------- depthwise causal conv4 + SiLU
__global__ __launch_bounds__(256)
void conv_silu_kernel(const float* __restrict__ XIN, const float* __restrict__ cw,
                      const float* __restrict__ cb, float* __restrict__ XC,
                      _Float16* __restrict__ XCH) {
  int m = blockIdx.x, d = threadIdx.x;
  int bb = m / L_, l = m % L_;
  float acc = cb[d];
#pragma unroll
  for (int j = 0; j < 4; j++) {
    int ll = l - 3 + j;
    if (ll >= 0) acc += cw[d * 4 + j] * XIN[((size_t)bb * L_ + ll) * DI_ + d];
  }
  float sv = acc / (1.f + expf(-acc));
  XC[(size_t)m * DI_ + d]  = sv;
  XCH[(size_t)m * DI_ + d] = (_Float16)sv;
}

// -------------------------------------------------- dt projection + softplus
__global__ __launch_bounds__(256)
void dtproj_kernel(const float* __restrict__ DBL, const float* __restrict__ dtw,
                   const float* __restrict__ dtb, float* __restrict__ DT) {
  int m = blockIdx.x, d = threadIdx.x;
  float acc = dtb[d];
#pragma unroll
  for (int r = 0; r < DTR_; r++)
    acc += DBL[(size_t)m * 40 + r] * dtw[d * DTR_ + r];
  float sp = (acc > 20.f) ? acc : log1pf(expf(acc));
  DT[(size_t)m * DI_ + d] = sp;
}

// ------------------------------- selective scan (recurrence over L) + gating
__global__ __launch_bounds__(256)
void scan_kernel(const float* __restrict__ DT, const float* __restrict__ XC,
                 const float* __restrict__ DBL, const _Float16* __restrict__ ZSH,
                 const float* __restrict__ A_log, const float* __restrict__ Dp,
                 _Float16* __restrict__ YH) {
  int bb = blockIdx.x;        // batch
  int d  = threadIdx.x;       // channel in DI
  float Arow[DS_];
#pragma unroll
  for (int s = 0; s < DS_; s++) Arow[s] = -expf(A_log[d * DS_ + s]);
  float dp = Dp[d];
  float h[DS_];
#pragma unroll
  for (int s = 0; s < DS_; s++) h[s] = 0.f;
  for (int l = 0; l < L_; l++) {
    size_t m = (size_t)bb * L_ + l;
    float dt = DT[m * DI_ + d];
    float u  = XC[m * DI_ + d];
    float du = dt * u;
    float y  = 0.f;
#pragma unroll
    for (int s = 0; s < DS_; s++) {
      float Bv = DBL[m * 40 + 8 + s];
      float Cv = DBL[m * 40 + 24 + s];
      h[s] = expf(dt * Arow[s]) * h[s] + du * Bv;
      y += h[s] * Cv;
    }
    y += u * dp;
    y *= (float)ZSH[m * DI_ + d];          // * silu(z), precomputed f16
    YH[m * DI_ + d] = (_Float16)y;
  }
}

// ------------------------------------------------------------------ launcher
extern "C" void kernel_launch(void* const* d_in, const int* in_sizes, int n_in,
                              void* d_out, int out_size, void* d_ws,
                              size_t ws_size, hipStream_t stream) {
  (void)in_sizes; (void)n_in; (void)out_size; (void)ws_size;
  const int*   ids      = (const int*)d_in[0];
  const float* item_emb = (const float*)d_in[1];
  const float* pos_emb  = (const float*)d_in[2];
  const float* ln0_w    = (const float*)d_in[3];
  const float* ln0_b    = (const float*)d_in[4];
  const float* in_w     = (const float*)d_in[5];
  const float* conv_w   = (const float*)d_in[6];
  const float* conv_b   = (const float*)d_in[7];
  const float* xproj_w  = (const float*)d_in[8];
  const float* dtproj_w = (const float*)d_in[9];
  const float* dtproj_b = (const float*)d_in[10];
  const float* A_log    = (const float*)d_in[11];
  const float* Dp       = (const float*)d_in[12];
  const float* out_w    = (const float*)d_in[13];
  const float* ln1_w    = (const float*)d_in[14];
  const float* ln1_b    = (const float*)d_in[15];
  const float* w1       = (const float*)d_in[16];
  const float* b1       = (const float*)d_in[17];
  const float* w2       = (const float*)d_in[18];
  const float* b2       = (const float*)d_in[19];
  const float* ln2_w    = (const float*)d_in[20];
  const float* ln2_b    = (const float*)d_in[21];

  char* ws = (char*)d_ws;
  size_t off = 0;
  auto alloc = [&](size_t bytes) {
    char* p = ws + off;
    off += (bytes + 255) & ~(size_t)255;
    return p;
  };
  float*    X    = (float*)   alloc((size_t)M_ * D_ * 4);
  _Float16* XH   = (_Float16*)alloc((size_t)M_ * D_ * 2);
  float*    XIN  = (float*)   alloc((size_t)M_ * DI_ * 4);  // aliased as HOUT
  _Float16* ZSH  = (_Float16*)alloc((size_t)M_ * DI_ * 2);  // silu(z) f16
  float*    XC   = (float*)   alloc((size_t)M_ * DI_ * 4);  // aliased as F1H
  _Float16* XCH  = (_Float16*)alloc((size_t)M_ * DI_ * 2);  // aliased as YH
  float*    DBL  = (float*)   alloc((size_t)M_ * 40 * 4);
  float*    DT   = (float*)   alloc((size_t)M_ * DI_ * 4);  // aliased as F2
  _Float16* in_wh  = (_Float16*)alloc((size_t)NL_ * 2 * DI_ * D_ * 2);
  _Float16* xprojh = (_Float16*)alloc((size_t)NL_ * 40 * DI_ * 2);
  _Float16* outwh  = (_Float16*)alloc((size_t)NL_ * D_ * DI_ * 2);
  _Float16* w1h    = (_Float16*)alloc((size_t)NL_ * FF_ * D_ * 2);
  _Float16* w2h    = (_Float16*)alloc((size_t)NL_ * D_ * FF_ * 2);

  float*    HOUT = XIN;              // free after conv consumes XIN
  _Float16* YH   = XCH;              // free after x_proj GEMM consumes XCH
  _Float16* F1H  = (_Float16*)XC;    // free after scan consumes XC (same bytes)
  float*    F2   = DT;               // free after scan consumes DT

  auto cvt = [&](const float* s, _Float16* dst, int n) {
    cvt_f16_kernel<<<(n + 255) / 256, 256, 0, stream>>>(s, dst, n);
  };
  cvt(in_w,    in_wh,  NL_ * 2 * DI_ * D_);
  cvt(xproj_w, xprojh, NL_ * 40 * DI_);
  cvt(out_w,   outwh,  NL_ * D_ * DI_);
  cvt(w1,      w1h,    NL_ * FF_ * D_);
  cvt(w2,      w2h,    NL_ * D_ * FF_);

  embed_ln0_kernel<<<M_ / 8, 256, 0, stream>>>(ids, item_emb, pos_emb, ln0_w,
                                               ln0_b, X, XH);

  for (int layer = 0; layer < NL_; layer++) {
    const _Float16* iwh = in_wh + (size_t)layer * 2 * DI_ * D_;
    // xz = x @ in_w^T, split halves: xin (f32) and silu(z) (f16)
    launch_gemm(EPI_F32, XH, iwh, nullptr, XIN, M_, DI_, D_, stream);
    launch_gemm(EPI_SILU_F16, XH, iwh + (size_t)DI_ * D_, nullptr, ZSH,
                M_, DI_, D_, stream);
    // depthwise causal conv4 + silu
    conv_silu_kernel<<<M_, 256, 0, stream>>>(XIN, conv_w + layer * DI_ * 4,
                                             conv_b + layer * DI_, XC, XCH);
    // dbl = xc @ xproj^T  (N=40, column-clamped)
    launch_gemm(EPI_F32, XCH, xprojh + (size_t)layer * 40 * DI_, nullptr, DBL,
                M_, 40, DI_, stream);
    // dt = softplus(dbl[:,:8] @ dtproj^T + b)
    dtproj_kernel<<<M_, 256, 0, stream>>>(DBL, dtproj_w + layer * DI_ * DTR_,
                                          dtproj_b + layer * DI_, DT);
    // selective scan + (y + u*Dp)*silu(z) -> f16
    scan_kernel<<<B_, 256, 0, stream>>>(DT, XC, DBL,
                                        ZSH, A_log + layer * DI_ * DS_,
                                        Dp + layer * DI_, YH);
    // out = y @ out_w^T
    launch_gemm(EPI_F32, YH, outwh + (size_t)layer * D_ * DI_, nullptr, HOUT,
                M_, D_, DI_, stream);
    // x = LN(h + x)
    add_ln_kernel<<<M_ / 8, 256, 0, stream>>>(HOUT, X, ln1_w + layer * D_,
                                              ln1_b + layer * D_, X, XH);
    // FFN: f1 = gelu(x @ w1^T + b1) (f16); f2 = f1 @ w2^T + b2 (f32)
    launch_gemm(EPI_GELU_BIAS_F16, XH, w1h + (size_t)layer * FF_ * D_,
                b1 + layer * FF_, F1H, M_, FF_, D_, stream);
    launch_gemm(EPI_BIAS_F32, F1H, w2h + (size_t)layer * D_ * FF_,
                b2 + layer * D_, F2, M_, D_, FF_, stream);
    // x = LN(f + x); final layer writes straight to d_out
    float* xdst = (layer == NL_ - 1) ? (float*)d_out : X;
    add_ln_kernel<<<M_ / 8, 256, 0, stream>>>(F2, X, ln2_w + layer * D_,
                                              ln2_b + layer * D_, xdst, XH);
  }
}
